// GINEMultiEdgeset_13666585935970
// MI455X (gfx1250) — matrix operand order repaired and checked by reading, not compiled, not run on hardware
//
#include <hip/hip_runtime.h>
#include <hip/hip_bf16.h>

#define NNODES 20000
#define NEDGES 320000
#define HD 128
#define RCRC 4
#define OUTD 64

typedef __attribute__((ext_vector_type(16))) __bf16 v16bf;
typedef __attribute__((ext_vector_type(8)))  float  v8f;

__device__ __forceinline__ unsigned short f2bf(float x) {
  unsigned int u = __float_as_uint(x);
  unsigned int r = u + 0x7FFFu + ((u >> 16) & 1u);   // round-to-nearest-even
  return (unsigned short)(r >> 16);
}

__device__ __forceinline__ float gelu_exact(float x) {
  return 0.5f * x * (1.0f + erff(x * 0.70710678118654752f));
}

// A fragment 16x32 bf16 (wave32): lanes 0-15 -> K {0..7,16..23}, lanes 16-31 -> K {8..15,24..31}
// Both halves are 16B-contiguous runs -> 2x ds_load_b128 per fragment.
__device__ __forceinline__ v16bf load_fragA(const unsigned short* p, int kk, int lane) {
  union { v16bf v; uint4 q[2]; } f;
  const int k0 = kk + ((lane >> 4) << 3);
  f.q[0] = *(const uint4*)(p + k0);
  f.q[1] = *(const uint4*)(p + k0 + 16);
  return f.v;
}

// B fragment 32x16 bf16 from transposed [N][K] LDS: lanes 0-15 -> K {0..7,16..23}, lanes 16-31 -> K {8..15,24..31}
// Two 16B-contiguous runs -> 2x ds_load_b128.
__device__ __forceinline__ v16bf load_fragB(const unsigned short* p, int kk, int lane) {
  union { v16bf v; uint4 q[2]; } f;
  const int k0 = kk + ((lane >> 4) << 3);
  f.q[0] = *(const uint4*)(p + k0);
  f.q[1] = *(const uint4*)(p + k0 + 16);
  return f.v;
}

// C[M x Nc] = act( op(A) @ W + bias ); op(A) = (1+eps)*A + A2 when A2 != null.
// Block = 256 threads = 8 waves; WN tiles along N, 8/WN tiles along M.
// Requires M % (16*(8/WN)) == 0. LDS rows padded to K+8 halves: 16B-aligned rows,
// non-power-of-two dword stride (bank spread), b128-friendly.
template <int K, int Nc, int WN>
__global__ __launch_bounds__(256) void wmma_gemm_kernel(
    const float* __restrict__ A, const float* __restrict__ A2,
    const float* __restrict__ epsPtr,
    const float* __restrict__ W, const float* __restrict__ bias,
    float* __restrict__ Cout,
    int doGelu, int reps, int repStride)
{
  constexpr int strideL = K + 8;
  constexpr int WM = 8 / WN;
  constexpr int blockRows = 16 * WM;
  __shared__ unsigned short ldsW[Nc * strideL];        // transposed [n][k]
  __shared__ unsigned short ldsA[blockRows * strideL]; // [r][k]
  const int tid = threadIdx.x;
  const int row0 = blockIdx.x * blockRows;

  for (int i = tid; i < K * Nc; i += 256) {
    const int k = i / Nc, n = i - k * Nc;
    ldsW[n * strideL + k] = f2bf(W[i]);
  }
  float scale = 1.0f;
  if (epsPtr) scale = 1.0f + epsPtr[0];
  for (int i = tid; i < blockRows * K; i += 256) {
    const int r = i / K, k = i - r * K;
    const int g = (row0 + r) * K + k;
    float vv = A[g];
    if (A2) vv = scale * vv + A2[g];
    ldsA[r * strideL + k] = f2bf(vv);
  }
  __syncthreads();

  const int wave = tid >> 5, lane = tid & 31;
  const int wm = wave / WN, wn = wave - wm * WN;
  const unsigned short* pa = ldsA + (wm * 16 + (lane & 15)) * strideL;
  const unsigned short* pb = ldsW + (wn * 16 + (lane & 15)) * strideL;

  v8f acc = {0.f, 0.f, 0.f, 0.f, 0.f, 0.f, 0.f, 0.f};
#pragma unroll
  for (int kk = 0; kk < K; kk += 32) {
    v16bf a = load_fragA(pa, kk, lane);
    v16bf b = load_fragB(pb, kk, lane);
    acc = __builtin_amdgcn_wmma_f32_16x16x32_bf16(false, a, false, b, (short)0, acc, false, false);
  }

  const int col = wn * 16 + (lane & 15);
  const float bv = bias ? bias[col] : 0.0f;
#pragma unroll
  for (int r = 0; r < 8; ++r) {
    const int m = ((lane >> 4) << 3) + r;       // C/D layout: vgpr r, lanes>=16 -> M+8
    const int row = row0 + wm * 16 + m;
    float val = acc[r] + bv;
    if (doGelu) val = gelu_exact(val);
    const int off = row * Nc + col;
    for (int rep = 0; rep < reps; ++rep) Cout[off + rep * repStride] = val;
  }
}

// msg = gelu(h[src] + edge_attr@bondW + bondb) * mask[src]*mask[dst]; atomic scatter-add into agg[dst].
// Block = 256 threads covers 8 edges of one rc replica (32 lanes x 4 features each).
// bondW (8KB, raw f32) staged via async global->LDS DMA (ASYNCcnt path).
__global__ __launch_bounds__(256) void msg_scatter_kernel(
    const float* __restrict__ h, const int* __restrict__ srcI, const int* __restrict__ dstI,
    const float* __restrict__ edge_attr, const float* __restrict__ bondW,
    const float* __restrict__ bondb, const float* __restrict__ node_mask,
    float* __restrict__ agg)
{
  __shared__ float sW[16 * 128];
  __shared__ float sB[128];
  __shared__ float sEA[8 * 16];
  const int tid = threadIdx.x;

  // async copy bondW -> sW: 512 x b128, 2 instructions per wave, no VGPR round-trip
  {
    const unsigned ldsBase = (unsigned)(unsigned long long)(void*)sW;
#pragma unroll
    for (int t = 0; t < 2; ++t) {
      const int i = tid + t * 256;                       // 16B chunk index, 0..511
      const unsigned laddr = ldsBase + (unsigned)(i * 16);
      const unsigned long long gaddr = (unsigned long long)(bondW + i * 4);
      asm volatile("global_load_async_to_lds_b128 %0, %1, off"
                   :: "v"(laddr), "v"(gaddr) : "memory");
    }
  }
  if (tid < 128) sB[tid] = bondb[tid];
  const int blocksPerRC = (NEDGES * 32) / 256;  // 40000
  const int rc = blockIdx.x / blocksPerRC;
  const int e0 = (blockIdx.x - rc * blocksPerRC) * 8;
  if (tid < 128) sEA[tid] = edge_attr[e0 * 16 + tid];
  asm volatile("s_wait_asynccnt 0" ::: "memory");
  __syncthreads();

  const int le = tid >> 5;
  const int e = e0 + le;
  const int lane = tid & 31;
  const int c4 = lane << 2;
  const int s = srcI[e], d = dstI[e];
  const float em = node_mask[rc * NNODES + s] * node_mask[rc * NNODES + d];
  const float4 h4 = *(const float4*)(h + (size_t)(rc * NNODES + s) * HD + c4);

  float ee[4];
#pragma unroll
  for (int j = 0; j < 4; ++j) ee[j] = sB[c4 + j];
#pragma unroll
  for (int k = 0; k < 16; ++k) {
    const float ea = sEA[le * 16 + k];
    const float* wrow = sW + k * 128 + c4;
#pragma unroll
    for (int j = 0; j < 4; ++j) ee[j] += ea * wrow[j];
  }

  float* ap = agg + (size_t)(rc * NNODES + d) * HD + c4;
  const float hv[4] = {h4.x, h4.y, h4.z, h4.w};
#pragma unroll
  for (int j = 0; j < 4; ++j) {
    const float msg = gelu_exact(hv[j] + ee[j]) * em;
    __hip_atomic_fetch_add(ap + j, msg, __ATOMIC_RELAXED, __HIP_MEMORY_SCOPE_AGENT);
  }
}

__device__ __forceinline__ float block128_sum(float v, float* red) {
#pragma unroll
  for (int off = 16; off >= 1; off >>= 1) v += __shfl_xor(v, off, 32);
  const int wave = threadIdx.x >> 5;
  if ((threadIdx.x & 31) == 0) red[wave] = v;
  __syncthreads();
  const float t = red[0] + red[1] + red[2] + red[3];
  __syncthreads();
  return t;
}

// h += gelu(LayerNorm(X) * g + b)   (one 128-thread block per row)
__global__ __launch_bounds__(128) void ln_gelu_res_kernel(
    const float* __restrict__ X, float* __restrict__ H,
    const float* __restrict__ g, const float* __restrict__ b)
{
  __shared__ float red[4];
  const int row = blockIdx.x, tid = threadIdx.x;
  const float x = X[row * HD + tid];
  const float mean = block128_sum(x, red) * (1.0f / HD);
  const float dx = x - mean;
  const float var = block128_sum(dx * dx, red) * (1.0f / HD);
  const float y = dx * rsqrtf(var + 1e-5f) * g[tid] + b[tid];
  H[row * HD + tid] += gelu_exact(y);
}

// X = relu(LayerNorm(X) * g + b)  in place
__global__ __launch_bounds__(128) void ln_relu_kernel(
    float* __restrict__ X, const float* __restrict__ g, const float* __restrict__ b)
{
  __shared__ float red[4];
  const int row = blockIdx.x, tid = threadIdx.x;
  const float x = X[row * HD + tid];
  const float mean = block128_sum(x, red) * (1.0f / HD);
  const float dx = x - mean;
  const float var = block128_sum(dx * dx, red) * (1.0f / HD);
  const float y = dx * rsqrtf(var + 1e-5f) * g[tid] + b[tid];
  X[row * HD + tid] = fmaxf(y, 0.0f);
}

// out[rc][o] = sum_n z2[rc,n,o]*mask[rc,n] / sum_n mask[rc,n]; one block per rc
__global__ __launch_bounds__(256) void pool_kernel(
    const float* __restrict__ z2, const float* __restrict__ mask, float* __restrict__ out)
{
  __shared__ float sred[256];
  __shared__ float mred[256];
  const int rc = blockIdx.x, tid = threadIdx.x;
  const int o = tid & 63, gq = tid >> 6;
  float s = 0.0f;
  for (int n = gq; n < NNODES; n += 4)
    s += z2[(size_t)(rc * NNODES + n) * OUTD + o] * mask[rc * NNODES + n];
  sred[tid] = s;
  float md = 0.0f;
  for (int n = tid; n < NNODES; n += 256) md += mask[rc * NNODES + n];
  mred[tid] = md;
  __syncthreads();
  for (int st = 128; st >= 1; st >>= 1) {
    if (tid < st) mred[tid] += mred[tid + st];
    __syncthreads();
  }
  if (tid < 64) {
    const float tot = sred[tid] + sred[tid + 64] + sred[tid + 128] + sred[tid + 192];
    out[rc * OUTD + tid] = tot / mred[0];
  }
}

__global__ __launch_bounds__(256) void fill_zero_kernel(float* __restrict__ p, int n4) {
  const int i = blockIdx.x * 256 + threadIdx.x;
  if (i < n4) ((float4*)p)[i] = make_float4(0.f, 0.f, 0.f, 0.f);
}

extern "C" void kernel_launch(void* const* d_in, const int* in_sizes, int n_in,
                              void* d_out, int out_size, void* d_ws, size_t ws_size,
                              hipStream_t stream) {
  const float* x         = (const float*)d_in[0];
  const int*   edge_idx  = (const int*)d_in[2];
  const float* edge_attr = (const float*)d_in[3];
  const float* node_mask = (const float*)d_in[4];
  const float* atom_W    = (const float*)d_in[5];
  const float* atom_b    = (const float*)d_in[6];
  const float* bond_W    = (const float*)d_in[7];
  const float* bond_b    = (const float*)d_in[8];
  const float* epsv      = (const float*)d_in[9];
  const float* conv_W1   = (const float*)d_in[10];
  const float* conv_b1   = (const float*)d_in[11];
  const float* conv_W2   = (const float*)d_in[12];
  const float* conv_b2   = (const float*)d_in[13];
  const float* norm_g    = (const float*)d_in[14];
  const float* norm_b    = (const float*)d_in[15];
  const float* mlp_W1    = (const float*)d_in[16];
  const float* mlp_b1    = (const float*)d_in[17];
  const float* mlp_ng    = (const float*)d_in[18];
  const float* mlp_nb    = (const float*)d_in[19];
  const float* mlp_W2    = (const float*)d_in[20];
  const float* mlp_b2    = (const float*)d_in[21];

  // workspace: h [4,20000,128] | S [4,20000,128] (agg -> mlp scratch, reused in place) | z2 [4,20000,64]
  float* h  = (float*)d_ws;
  float* S  = h + (size_t)RCRC * NNODES * HD;
  float* z2 = S + (size_t)RCRC * NNODES * HD;

  const int* srcI = edge_idx;
  const int* dstI = edge_idx + NEDGES;
  const int rowsAll = RCRC * NNODES;          // 80000
  const int stateN4 = RCRC * NNODES * HD / 4; // for zero-fill

  // atom encoder: h0 = x@atom_W + atom_b, replicated to all 4 (r,c) replicas
  wmma_gemm_kernel<64, 128, 8><<<NNODES / 16, 256, 0, stream>>>(
      x, nullptr, nullptr, atom_W, atom_b, h, 0, RCRC, NNODES * HD);

  for (int l = 0; l < 4; ++l) {
    fill_zero_kernel<<<(stateN4 + 255) / 256, 256, 0, stream>>>(S, stateN4);
    msg_scatter_kernel<<<RCRC * (NEDGES * 32 / 256), 256, 0, stream>>>(
        h, srcI, dstI, edge_attr, bond_W + l * 16 * HD, bond_b + l * HD, node_mask, S);
    // t = gelu( ((1+eps)h + agg) @ W1 + b1 )   (in place into S; blocks only touch own rows)
    wmma_gemm_kernel<128, 128, 8><<<rowsAll / 16, 256, 0, stream>>>(
        h, S, epsv + l, conv_W1 + l * HD * HD, conv_b1 + l * HD, S, 1, 1, 0);
    // t = t @ W2 + b2  (in place)
    wmma_gemm_kernel<128, 128, 8><<<rowsAll / 16, 256, 0, stream>>>(
        S, nullptr, nullptr, conv_W2 + l * HD * HD, conv_b2 + l * HD, S, 0, 1, 0);
    // h += gelu(LN(t))
    ln_gelu_res_kernel<<<rowsAll, 128, 0, stream>>>(S, h, norm_g + l * HD, norm_b + l * HD);
  }

  // final head
  wmma_gemm_kernel<128, 128, 8><<<rowsAll / 16, 256, 0, stream>>>(
      h, nullptr, nullptr, mlp_W1, mlp_b1, S, 0, 1, 0);
  ln_relu_kernel<<<rowsAll, 128, 0, stream>>>(S, mlp_ng, mlp_nb);
  wmma_gemm_kernel<128, 64, 4><<<rowsAll / 32, 256, 0, stream>>>(
      S, nullptr, nullptr, mlp_W2, mlp_b2, z2, 0, 1, 0);
  pool_kernel<<<RCRC, 256, 0, stream>>>(z2, node_mask, (float*)d_out);
}
